// GCGC_32006096290301
// MI455X (gfx1250) — matrix operand order repaired and checked by reading, not compiled, not run on hardware
//
#include <hip/hip_runtime.h>
#include <hip/hip_bf16.h>

typedef __attribute__((ext_vector_type(16))) __bf16 v16bf;
typedef __attribute__((ext_vector_type(8)))  float  v8f;

struct Q2 { uint4 a, b; };               // 32B: raw bits of a v16bf fragment

#define B_  64
#define C_  64
#define T_  256
#define V_  25
#define S_  9
#define O_  256
#define VP_ 32
#define WSTR 40                          // padded LDS row stride (bf16 elems)
#define ZO_SIZE  104857600ull            // B*O*T*V
#define LOSS_DEN 26214400.0f             // B*V*C*T

#define ZB_BYTES 67108864ull             // B*C*T*VP bf16
#define XS_BYTES 409600ull               // B*C*V f32
#define WF_BYTES 1179648ull              // B*S*VP*VP bf16

// ---------------------------------------------------------------- kernel 1
// xs[b,c,v] = sum_t x[b,c,t,v]   (conflict-free: lane owns v, 8 waves split t)
__global__ __launch_bounds__(256) void k_reduce_t(const float* __restrict__ x,
                                                  float* __restrict__ xs) {
  __shared__ float part[8][32];
  int blk = blockIdx.x;                  // b*C + c
  int tid = threadIdx.x;
  int w = tid >> 5, lane = tid & 31;
  const float* xp = x + (size_t)blk * (T_ * V_);
  float acc = 0.f;
  if (lane < V_) {
    for (int t = w; t < T_; t += 8) acc += xp[(size_t)t * V_ + lane];
  }
  part[w][lane] = acc;
  __syncthreads();
  if (tid < V_) {
    float s = 0.f;
    for (int j = 0; j < 8; ++j) s += part[j][tid];
    xs[(size_t)blk * V_ + tid] = s;
  }
}

// ---------------------------------------------------------------- kernel 2
// per b: x1m/x2m -> rel -> ridge (to d_out)
// Wf stored u-major: Wf[b][k][u32][v32] = rel[b,8-k,v,u] + A[v,u]  (pads = 0)
__global__ __launch_bounds__(256) void k_prep(const float* __restrict__ xs,
    const float* __restrict__ Ap, const float* __restrict__ w1,
    const float* __restrict__ b1, const float* __restrict__ w2,
    const float* __restrict__ b2, unsigned short* __restrict__ Wf_,
    float* __restrict__ out, float* __restrict__ lossp) {
  __bf16* Wf = reinterpret_cast<__bf16*>(Wf_);
  __shared__ float sx1[S_ * V_], sx2[S_ * V_], red[256];
  int b = blockIdx.x, tid = threadIdx.x;
  if (b == 0 && tid == 0) lossp[0] = 0.f;   // reset loss accumulator each call
  if (tid < S_ * V_) {
    int s = tid / V_, v = tid % V_;
    float a1 = 0.f, a2 = 0.f;
    for (int c = 0; c < C_; ++c) {
      float xv = xs[((size_t)b * C_ + c) * V_ + v];
      a1 += xv * w1[s * C_ + c];
      a2 += xv * w2[s * C_ + c];
    }
    sx1[tid] = a1 * (1.f / T_) + b1[s];
    sx2[tid] = a2 * (1.f / T_) + b2[s];
  }
  __syncthreads();
  float rsum = 0.f;
  __bf16* Wfb = Wf + (size_t)b * (S_ * VP_ * VP_);
  for (int i = tid; i < S_ * VP_ * VP_; i += 256) {
    int k = i >> 10, rem = i & 1023, u = rem >> 5, v = rem & 31;  // u-major
    float w = 0.f;
    if (v < V_ && u < V_) {
      int s = (S_ - 1) - k;                     // W[k] = rel[8-k]
      float r = tanhf(sx1[s * V_ + v] - sx2[s * V_ + u]);
      rsum += r * r;
      w = r + Ap[v * V_ + u];                   // ALPHA == 1
    }
    Wfb[i] = (__bf16)w;
  }
  red[tid] = rsum;
  __syncthreads();
  for (int s2 = 128; s2 > 0; s2 >>= 1) {
    if (tid < s2) red[tid] += red[tid + s2];
    __syncthreads();
  }
  if (tid == 0) out[ZO_SIZE + 1 + b] = red[0];
}

// ---------------------------------------------------------------- kernel 3
// z[b,c,t,u] = sum_k sum_v x[b,c,t-k,v] * W[b,k,v,u]  (9 shifted bf16 WMMAs)
// stores zB[b][c][t*32+u] bf16 via LDS staging (coalesced b128 stores);
// accumulates loss = sum (z - x)^2 over u<25.
__global__ __launch_bounds__(256) void k_conv(const float* __restrict__ x,
    const unsigned short* __restrict__ Wf_, unsigned short* __restrict__ zB_,
    float* __restrict__ lossp) {
  const __bf16* Wf = reinterpret_cast<const __bf16*>(Wf_);
  __bf16* zB = reinterpret_cast<__bf16*>(zB_);
  __shared__ __align__(16) __bf16 Xb[136 * WSTR];      // x tile, rows t0-8..t0+127
  __shared__ __align__(16) __bf16 Wl[S_ * VP_ * WSTR]; // [k][u][v], stride 40
  __shared__ __align__(16) __bf16 zstage[128 * VP_];   // [t_local][u]
  __shared__ float red[256];
  int blk = blockIdx.x;
  int tb = blk & 1, c = (blk >> 1) & 63, b = blk >> 7;
  int tid = threadIdx.x;
  int t0 = tb * 128;

  const __bf16* wsrc = Wf + (size_t)b * (S_ * VP_ * VP_);
  for (int i = tid; i < S_ * VP_ * VP_; i += 256) {
    int ku = i >> 5, v = i & 31;                 // ku = k*32+u
    Wl[ku * WSTR + v] = wsrc[i];
  }

  const float* xp = x + (((size_t)b * C_ + c) * T_) * V_;
  __builtin_prefetch(xp + (size_t)t0 * V_, 0, 1);   // global_prefetch_b8
  for (int i = tid; i < 136 * VP_; i += 256) {
    int row = i >> 5, vv = i & 31;
    int t = t0 - (S_ - 1) + row;
    float val = 0.f;
    if (vv < V_ && t >= 0) val = xp[(size_t)t * V_ + vv];   // causal zero-pad
    Xb[row * WSTR + vv] = (__bf16)val;
  }
  __syncthreads();

  int w = tid >> 5, lane = tid & 31, m = lane & 15;
  bool lh = lane < 16;
  int kb = lh ? 0 : 8;                    // A: K split {0..7,16..23}/{8..15,24..31}
  int vb = lh ? 0 : 16;                   // unused; B loads the same K split as A (via kb)
  int mbase = lh ? 0 : 8;
  float lsum = 0.f;

  for (int ut = 0; ut < 2; ++ut) {
    v8f acc = {};
    int u = ut * 16 + m;
#pragma unroll
    for (int k = 0; k < S_; ++k) {
      int row = 16 * w + m + (S_ - 1) - k;
      const __bf16* xr = &Xb[row * WSTR];
      Q2 qa;
      qa.a = *reinterpret_cast<const uint4*>(xr + kb);        // K kb..kb+7
      qa.b = *reinterpret_cast<const uint4*>(xr + kb + 16);   // K kb+16..kb+23
      v16bf a = __builtin_bit_cast(v16bf, qa);
      const __bf16* wr = &Wl[(k * VP_ + u) * WSTR + kb];
      Q2 qb;
      qb.a = *reinterpret_cast<const uint4*>(wr);
      qb.b = *reinterpret_cast<const uint4*>(wr + 16);
      v16bf fb = __builtin_bit_cast(v16bf, qb);
      acc = __builtin_amdgcn_wmma_f32_16x16x32_bf16(false, a, false, fb,
                                                    (short)0, acc, false, false);
    }
#pragma unroll
    for (int r = 0; r < 8; ++r) {
      int tl = 16 * w + r + mbase;
      float val = acc[r];
      zstage[tl * VP_ + u] = (__bf16)val;
      if (u < V_) {
        float d = val - (float)Xb[(tl + (S_ - 1)) * WSTR + u];
        lsum += d * d;
      }
    }
  }
  red[tid] = lsum;
  __syncthreads();
  // coalesced flush: 128*32 bf16 = 8 KB = 512 uint4
  {
    const uint4* zsv = reinterpret_cast<const uint4*>(zstage);
    uint4* gdst = reinterpret_cast<uint4*>(
        zB + (((size_t)b * C_ + c) * T_ + t0) * VP_);
    for (int i = tid; i < 512; i += 256) gdst[i] = zsv[i];
  }
  for (int s2 = 128; s2 > 0; s2 >>= 1) {
    if (tid < s2) red[tid] += red[tid + s2];
    __syncthreads();
  }
  if (tid == 0) atomicAdd(lossp, red[0]);
}

// ---------------------------------------------------------------- kernel 4
// zo[b,o,t,v] = sum_c z[b,c,t,v]*w3[o,c] + b3[o]  (A=w3, B=z cols, K=64)
// zB global layout [b][c][n], n = t*32+u; transposed to [n][c] in LDS.
__global__ __launch_bounds__(256) void k_out(const unsigned short* __restrict__ zB_,
    const float* __restrict__ w3, const float* __restrict__ b3,
    float* __restrict__ out, const float* __restrict__ lossp) {
  __shared__ __align__(16) __bf16 w3l[O_ * 72];  // row stride 72 (144B, 16B-align)
  __shared__ __align__(16) __bf16 zl[128 * 72];  // [n_local][c], stride 72
  __shared__ float b3l[O_];
  unsigned short* zl16 = reinterpret_cast<unsigned short*>(zl);
  int blk = blockIdx.x;
  int chunk = blk & 63, b = blk >> 6;
  int tid = threadIdx.x;
  int n0 = chunk * 128;                   // n = t*32 + u  (columns)

  for (int i = tid; i < O_ * C_; i += 256)
    w3l[(i >> 6) * 72 + (i & 63)] = (__bf16)w3[i];
  if (tid < O_) b3l[tid] = b3[tid];
  // coalesced dword loads from [b][c][n], transpose-store into LDS [n][c]
  {
    const unsigned short* zsrc = zB_ + (size_t)b * C_ * (T_ * VP_);
    for (int i = tid; i < 64 * 64; i += 256) {      // 64 c * 64 dwords
      int c = i >> 6, n2 = (i & 63) * 2;
      unsigned int val =
          *reinterpret_cast<const unsigned int*>(zsrc + (size_t)c * (T_ * VP_) + n0 + n2);
      zl16[n2 * 72 + c]       = (unsigned short)(val & 0xffffu);
      zl16[(n2 + 1) * 72 + c] = (unsigned short)(val >> 16);
    }
  }
  __syncthreads();

  if (blk == 0 && tid == 0) out[ZO_SIZE] = lossp[0] * (1.f / LOSS_DEN);

  int w = tid >> 5, lane = tid & 31, m = lane & 15;
  bool lh = lane < 16;
  int kb = lh ? 0 : 8;
  int koff = lh ? 0 : 16;
  int mbase = lh ? 0 : 8;

  for (int oi = 0; oi < 2; ++oi) {
    int ot = w * 2 + oi;
    int o = ot * 16 + m;
    const __bf16* wr = &w3l[o * 72];
    Q2 q;
    q.a = *reinterpret_cast<const uint4*>(wr + kb);
    q.b = *reinterpret_cast<const uint4*>(wr + kb + 16);
    v16bf a0 = __builtin_bit_cast(v16bf, q);
    q.a = *reinterpret_cast<const uint4*>(wr + 32 + kb);
    q.b = *reinterpret_cast<const uint4*>(wr + 48 + kb);
    v16bf a1 = __builtin_bit_cast(v16bf, q);
    for (int nt = 0; nt < 8; ++nt) {
      const __bf16* zr = &zl[(nt * 16 + m) * 72 + kb];
      Q2 qb;
      qb.a = *reinterpret_cast<const uint4*>(zr);
      qb.b = *reinterpret_cast<const uint4*>(zr + 16);
      v16bf fb0 = __builtin_bit_cast(v16bf, qb);
      qb.a = *reinterpret_cast<const uint4*>(zr + 32);
      qb.b = *reinterpret_cast<const uint4*>(zr + 48);
      v16bf fb1 = __builtin_bit_cast(v16bf, qb);
      v8f acc = {};
      acc = __builtin_amdgcn_wmma_f32_16x16x32_bf16(false, a0, false, fb0,
                                                    (short)0, acc, false, false);
      acc = __builtin_amdgcn_wmma_f32_16x16x32_bf16(false, a1, false, fb1,
                                                    (short)0, acc, false, false);
      int n = n0 + nt * 16 + m;
      int t = n >> 5, u = n & 31;
      if (u < V_) {
#pragma unroll
        for (int r = 0; r < 8; ++r) {
          int o2 = ot * 16 + r + mbase;
          out[((size_t)(b * O_ + o2) * T_ + t) * V_ + u] = acc[r] + b3l[o2];
        }
      }
    }
  }
}

// ---------------------------------------------------------------- launcher
extern "C" void kernel_launch(void* const* d_in, const int* in_sizes, int n_in,
                              void* d_out, int out_size, void* d_ws, size_t ws_size,
                              hipStream_t stream) {
  (void)in_sizes; (void)n_in; (void)out_size; (void)ws_size;
  const float* x  = (const float*)d_in[0];
  const float* Ap = (const float*)d_in[1];
  const float* w1 = (const float*)d_in[2];
  const float* b1 = (const float*)d_in[3];
  const float* w2 = (const float*)d_in[4];
  const float* b2 = (const float*)d_in[5];
  const float* w3 = (const float*)d_in[6];
  const float* b3 = (const float*)d_in[7];
  float* out = (float*)d_out;

  char* ws = (char*)d_ws;
  unsigned short* zB = (unsigned short*)ws;
  float* xs          = (float*)(ws + ZB_BYTES);
  unsigned short* Wf = (unsigned short*)(ws + ZB_BYTES + XS_BYTES);
  float* lossp       = (float*)(ws + ZB_BYTES + XS_BYTES + WF_BYTES);

  hipLaunchKernelGGL(k_reduce_t, dim3(B_ * C_), dim3(256), 0, stream, x, xs);
  hipLaunchKernelGGL(k_prep, dim3(B_), dim3(256), 0, stream,
                     xs, Ap, w1, b1, w2, b2, Wf, out, lossp);
  hipLaunchKernelGGL(k_conv, dim3(B_ * C_ * 2), dim3(256), 0, stream,
                     x, Wf, zB, lossp);
  hipLaunchKernelGGL(k_out, dim3(B_ * 64), dim3(256), 0, stream,
                     zB, w3, b3, out, lossp);
}
